// BernNet_NC_43542378447161
// MI455X (gfx1250) — compile-verified, matrix operand order checked by reading, not run on hardware
//
#include <hip/hip_runtime.h>
#include <hip/hip_bf16.h>
#include <math.h>

typedef __attribute__((ext_vector_type(16))) _Float16 v16h;
typedef __attribute__((ext_vector_type(4)))  _Float16 v4h;
typedef __attribute__((ext_vector_type(8)))  float    v8f;

#define N_NODES 100000
#define N_EDGES 3200000
#define N_FEATS 512
#define HIDDEN  256
#define N_CLS   40
#define N_CLSP  48   // padded to 3 WMMA tiles
#define KORD    10

// ---------------------------------------------------------------- utilities
__device__ __forceinline__ void atomAddF(float* p, float v) {
    unsafeAtomicAdd(p, v);   // lowers to global_atomic_add_f32 on gfx1250
}

__global__ void k_zero(float* __restrict__ buf, int n) {
    int i = blockIdx.x * blockDim.x + threadIdx.x;
    if (i < n) buf[i] = 0.0f;
}

// ---------------------------------------------------------------- weight prep
// W1 [512,256] f32 -> W1t [256,512] f16 (column-major for WMMA B-fragments)
__global__ void k_cast_w1t(const float* __restrict__ W1, _Float16* __restrict__ w1t) {
    int o = blockIdx.x * blockDim.x + threadIdx.x;
    if (o >= HIDDEN * N_FEATS) return;
    int n = o / N_FEATS, k = o - n * N_FEATS;
    w1t[o] = (_Float16)W1[k * HIDDEN + n];
}

// W2 [256,40] f32 -> W2t [48,256] f16, zero-padded cols 40..47
__global__ void k_cast_w2t(const float* __restrict__ W2, _Float16* __restrict__ w2t) {
    int o = blockIdx.x * blockDim.x + threadIdx.x;
    if (o >= N_CLSP * HIDDEN) return;
    int n = o / HIDDEN, k = o - n * HIDDEN;
    w2t[o] = (n < N_CLS) ? (_Float16)W2[k * N_CLS + n] : (_Float16)0.0f;
}

// Bernstein coefficients: c_j = C(K,j)/2^K * relu(temp[j])
__global__ void k_coef(const float* __restrict__ temp, float* __restrict__ cbuf) {
    const float comb[KORD + 1] = {1.f,10.f,45.f,120.f,210.f,252.f,210.f,120.f,45.f,10.f,1.f};
    int j = threadIdx.x;
    if (j <= KORD) cbuf[j] = comb[j] * (1.0f / 1024.0f) * fmaxf(temp[j], 0.0f);
}

// ---------------------------------------------------------------- graph prep
__global__ void k_deg(const long long* __restrict__ ei, float* __restrict__ deg) {
    int e = blockIdx.x * blockDim.x + threadIdx.x;
    if (e >= N_EDGES) return;
    atomAddF(&deg[(int)ei[e]], 1.0f);
}

__global__ void k_dinv(const float* __restrict__ deg, float* __restrict__ dinv) {
    int i = blockIdx.x * blockDim.x + threadIdx.x;
    if (i >= N_NODES) return;
    float d = deg[i];
    dinv[i] = (d > 0.0f) ? rsqrtf(d) : 0.0f;
}

__global__ void k_aw(const long long* __restrict__ ei, const float* __restrict__ dinv,
                     float* __restrict__ aw) {
    int e = blockIdx.x * blockDim.x + threadIdx.x;
    if (e >= N_EDGES) return;
    int s = (int)ei[e], d = (int)ei[N_EDGES + e];
    aw[e] = dinv[s] * dinv[d];
}

// ---------------------------------------------------------------- fused MLP (WMMA)
// block = 2 waves; each wave computes z for 16 nodes.  h never hits HBM.
__global__ __launch_bounds__(64) void k_mlp(const float* __restrict__ x,
                                            const _Float16* __restrict__ w1t,
                                            const float* __restrict__ b1,
                                            const _Float16* __restrict__ w2t,
                                            const float* __restrict__ b2,
                                            float* __restrict__ z) {
    __shared__ _Float16 xt[2][16 * N_FEATS];  // 32 KB
    __shared__ _Float16 ht[2][16 * HIDDEN];   // 16 KB
    const int lane = threadIdx.x & 31;
    const int w    = threadIdx.x >> 5;
    const int node_base = (blockIdx.x * 2 + w) * 16;
    const int half = lane >> 4;
    const int l15  = lane & 15;

    // stage x tile (16 nodes x 512 feats) into LDS as f16
    for (int it = 0; it < 64; ++it) {
        int flat = it * 32 + lane;          // float4 groups: 16*512/4 = 2048
        int row = flat >> 7, g = flat & 127;
        const float4 xv = *(const float4*)(x + (size_t)(node_base + row) * N_FEATS + g * 4);
        v4h hv = { (_Float16)xv.x, (_Float16)xv.y, (_Float16)xv.z, (_Float16)xv.w };
        *(v4h*)(&xt[w][row * N_FEATS + g * 4]) = hv;
    }
    __syncthreads();

    // GEMM1: h[16,256] = relu(x@W1 + b1) -> LDS (f16)
    for (int nt = 0; nt < 16; ++nt) {
        v8f acc = {};
        const int n = nt * 16 + l15;
        #pragma unroll
        for (int kk = 0; kk < 16; ++kk) {
            int kb = kk * 32 + half * 16;
            v16h a = *(const v16h*)(&xt[w][l15 * N_FEATS + kb]);
            v16h b = *(const v16h*)(w1t + (size_t)n * N_FEATS + kb);
            acc = __builtin_amdgcn_wmma_f32_16x16x32_f16(false, a, false, b,
                                                         (short)0, acc, false, false);
        }
        float bv = b1[n];
        #pragma unroll
        for (int v = 0; v < 8; ++v) {
            int m = v + half * 8;
            ht[w][m * HIDDEN + n] = (_Float16)fmaxf(acc[v] + bv, 0.0f);
        }
    }
    __syncthreads();

    // GEMM2: z[16,40] = h@W2 + b2 (N padded to 48)
    for (int nt = 0; nt < 3; ++nt) {
        v8f acc = {};
        const int n = nt * 16 + l15;
        #pragma unroll
        for (int kk = 0; kk < 8; ++kk) {
            int kb = kk * 32 + half * 16;
            v16h a = *(const v16h*)(&ht[w][l15 * HIDDEN + kb]);
            v16h b = *(const v16h*)(w2t + (size_t)n * HIDDEN + kb);
            acc = __builtin_amdgcn_wmma_f32_16x16x32_f16(false, a, false, b,
                                                         (short)0, acc, false, false);
        }
        if (n < N_CLS) {
            float bv = b2[n];
            #pragma unroll
            for (int v = 0; v < 8; ++v) {
                int m = v + half * 8;
                z[(size_t)(node_base + m) * N_CLS + n] = acc[v] + bv;
            }
        }
    }
}

// ---------------------------------------------------------------- propagation
__global__ void k_init_pr(const float* __restrict__ z, const float* __restrict__ cbuf,
                          float* __restrict__ p, float* __restrict__ r) {
    int i = blockIdx.x * blockDim.x + threadIdx.x;
    if (i >= N_NODES * N_CLS) return;
    float zz = z[i];
    p[i] = zz;
    r[i] = cbuf[KORD] * zz;
}

// t[dst,c] += a_w[e] * v[src,c]; 40 lanes per edge (vector gather is contiguous)
__global__ void k_spmv(float* __restrict__ t, const float* __restrict__ v,
                       const float* __restrict__ aw, const long long* __restrict__ ei) {
    unsigned tid = blockIdx.x * blockDim.x + threadIdx.x;
    unsigned e = tid / 40u;
    unsigned c = tid - e * 40u;
    if (e >= N_EDGES) return;
    int s = (int)ei[e], d = (int)ei[N_EDGES + e];
    float val = aw[e] * v[(size_t)s * N_CLS + c];
    atomAddF(&t[(size_t)d * N_CLS + c], val);
}

__global__ void k_padd(float* __restrict__ p, const float* __restrict__ t) {
    int i = blockIdx.x * blockDim.x + threadIdx.x;
    if (i < N_NODES * N_CLS) p[i] += t[i];
}

__global__ void k_rupd(float* __restrict__ r, const float* __restrict__ t,
                       const float* __restrict__ p, const float* __restrict__ cbuf, int j) {
    int i = blockIdx.x * blockDim.x + threadIdx.x;
    if (i >= N_NODES * N_CLS) return;
    r[i] = r[i] - t[i] + cbuf[j] * p[i];
}

// ---------------------------------------------------------------- log-softmax (wave/node)
__global__ void k_logsm(const float* __restrict__ r, float* __restrict__ out) {
    int lane = threadIdx.x & 31;
    int node = blockIdx.x * 8 + (threadIdx.x >> 5);
    const float* row = r + (size_t)node * N_CLS;
    float v0 = row[lane];
    float v1 = (lane < 8) ? row[lane + 32] : -INFINITY;
    float m = fmaxf(v0, v1);
    #pragma unroll
    for (int off = 16; off >= 1; off >>= 1) m = fmaxf(m, __shfl_xor(m, off));
    float s = __expf(v0 - m) + ((lane < 8) ? __expf(v1 - m) : 0.0f);
    #pragma unroll
    for (int off = 16; off >= 1; off >>= 1) s += __shfl_xor(s, off);
    float ls = __logf(s);
    float* orow = out + (size_t)node * N_CLS;
    orow[lane] = v0 - m - ls;
    if (lane < 8) orow[lane + 32] = v1 - m - ls;
}

// ---------------------------------------------------------------- host
extern "C" void kernel_launch(void* const* d_in, const int* in_sizes, int n_in,
                              void* d_out, int out_size, void* d_ws, size_t ws_size,
                              hipStream_t stream) {
    const float*     x    = (const float*)d_in[0];
    const long long* ei   = (const long long*)d_in[1];
    const float*     W1   = (const float*)d_in[2];
    const float*     b1   = (const float*)d_in[3];
    const float*     W2   = (const float*)d_in[4];
    const float*     b2   = (const float*)d_in[5];
    const float*     temp = (const float*)d_in[6];
    float*           out  = (float*)d_out;

    char* ws = (char*)d_ws;
    size_t o = 0;
    auto take = [&](size_t bytes) { size_t r = o; o = (o + bytes + 255) & ~(size_t)255; return r; };
    _Float16* w1t  = (_Float16*)(ws + take((size_t)HIDDEN * N_FEATS * 2));
    _Float16* w2t  = (_Float16*)(ws + take((size_t)N_CLSP * HIDDEN * 2));
    float*    cbuf = (float*)(ws + take((KORD + 1) * 4));
    float*    deg  = (float*)(ws + take((size_t)N_NODES * 4));
    float*    dinv = (float*)(ws + take((size_t)N_NODES * 4));
    float*    aw   = (float*)(ws + take((size_t)N_EDGES * 4));
    float*    z    = (float*)(ws + take((size_t)N_NODES * N_CLS * 4));
    float*    p    = (float*)(ws + take((size_t)N_NODES * N_CLS * 4));
    float*    rr   = (float*)(ws + take((size_t)N_NODES * N_CLS * 4));
    float*    t    = (float*)(ws + take((size_t)N_NODES * N_CLS * 4));

    const int T = 256;
    const int gNodes = (N_NODES + T - 1) / T;
    const int gEdges = (N_EDGES + T - 1) / T;
    const int gVec   = (N_NODES * N_CLS + T - 1) / T;
    const int gSpmv  = (int)(((long long)N_EDGES * 40 + T - 1) / T);

    // weight prep + coefficients
    k_cast_w1t<<<(HIDDEN * N_FEATS + T - 1) / T, T, 0, stream>>>(W1, w1t);
    k_cast_w2t<<<(N_CLSP * HIDDEN + T - 1) / T, T, 0, stream>>>(W2, w2t);
    k_coef<<<1, 32, 0, stream>>>(temp, cbuf);

    // graph normalization
    k_zero<<<gNodes, T, 0, stream>>>(deg, N_NODES);
    k_deg<<<gEdges, T, 0, stream>>>(ei, deg);
    k_dinv<<<gNodes, T, 0, stream>>>(deg, dinv);
    k_aw<<<gEdges, T, 0, stream>>>(ei, dinv, aw);

    // fused MLP (6250 node tiles, 2 tiles per block)
    k_mlp<<<N_NODES / 32, 64, 0, stream>>>(x, w1t, b1, w2t, b2, z);

    // Horner Bernstein propagation: 20 SpMVs instead of 65
    k_init_pr<<<gVec, T, 0, stream>>>(z, cbuf, p, rr);
    for (int j = KORD - 1; j >= 0; --j) {
        k_zero<<<gVec, T, 0, stream>>>(t, N_NODES * N_CLS);
        k_spmv<<<gSpmv, T, 0, stream>>>(t, p, aw, ei);
        k_padd<<<gVec, T, 0, stream>>>(p, t);           // p = (I+A)p
        k_zero<<<gVec, T, 0, stream>>>(t, N_NODES * N_CLS);
        k_spmv<<<gSpmv, T, 0, stream>>>(t, rr, aw, ei);
        k_rupd<<<gVec, T, 0, stream>>>(rr, t, p, cbuf, j); // r = (I-A)r + c_j p
    }

    // log_softmax
    k_logsm<<<N_NODES / 8, T, 0, stream>>>(rr, out);
}
